// EarthSpecificBlock_64828236365995
// MI455X (gfx1250) — compile-verified, matrix operand order checked by reading, not run on hardware
//
#include <hip/hip_runtime.h>
#include <hip/hip_bf16.h>
#include <math.h>

typedef __attribute__((ext_vector_type(16))) _Float16 v16h;
typedef __attribute__((ext_vector_type(8)))  _Float16 v8h;
typedef __attribute__((ext_vector_type(8)))  float    v8f;

#define WMMA_F32_F16(A,B,C) \
  __builtin_amdgcn_wmma_f32_16x16x32_f16(false,(A),false,(B),(short)0,(C),false,false)

// ---------------- static config (matches reference) ----------------
constexpr int DIM   = 192;
constexpr int HEADS = 6;
constexpr int HD    = 32;
constexpr int PLc = 8, LATc = 96, LONc = 180;
constexpr int WPc = 2, WLc = 6, WWc = 12;
constexpr int Nt  = 144;                 // tokens per window
constexpr int NLONc = 15;
constexpr int NWc  = 64;                 // type_of_windows
constexpr int NWIN = NLONc * NWc;        // 960 windows
constexpr int LTOK = PLc * LATc * LONc;  // 138240 tokens
constexpr float SCALE = 0.17677669529663687f;   // HD^-0.5

// ---------------- workspace layout (bytes, 256-aligned) ----------------
constexpr size_t SZ_XW  = (size_t)LTOK * DIM * 2;        // xw / attn_out, f16
constexpr size_t SZ_QKV = (size_t)LTOK * 3 * DIM * 2;    // qkv f16; later h2
constexpr size_t SZ_X1  = (size_t)LTOK * DIM * 4;        // x1 fp32
constexpr size_t SZ_ACT = (size_t)LTOK * 4 * DIM * 2;    // mlp act f16
constexpr size_t OFF_QKV = SZ_XW;
constexpr size_t OFF_X1  = OFF_QKV + SZ_QKV;
constexpr size_t OFF_ACT = OFF_X1 + SZ_X1;
constexpr size_t OFF_W   = OFF_ACT + SZ_ACT;

// ---------------- CDNA5 async copy (global -> LDS, ASYNCcnt) ----------------
// ISA 15.18.3: GLOBAL_LOAD_ASYNC_TO_LDS_B128, GV mode (64-bit vaddr, saddr=off).
// VDST VGPR holds the LDS byte address (generic LDS ptr low 32 bits).
__device__ __forceinline__ void async_ld128(unsigned lds_addr, const _Float16* gptr) {
  asm volatile("global_load_async_to_lds_b128 %0, %1, off"
               :: "v"(lds_addr), "v"((unsigned long long)(uintptr_t)gptr)
               : "memory");
}
template <int N>
__device__ __forceinline__ void wait_asynccnt() {
  asm volatile("s_wait_asynccnt %0" :: "n"(N) : "memory");
}

// ---------------- helpers ----------------

// window-ordered token t -> original flat token index (undoes roll + partition)
__device__ __forceinline__ int win_to_orig(int t) {
  int widx = t / Nt;
  int n    = t - widx * Nt;
  int lonw = widx / NWc;
  int w    = widx - lonw * NWc;
  int npl  = w >> 4;
  int nlat = w & 15;
  int wp  = n / (WLc * WWc);
  int rem = n - wp * (WLc * WWc);
  int wl  = rem / WWc;
  int ww  = rem - wl * WWc;
  int pl  = (npl * WPc + wp + 1) & 7;          // roll back shift +SP (mod 8)
  int lat = nlat * WLc + wl + 3; if (lat >= LATc) lat -= LATc;
  int lon = lonw * WWc + ww + 6; if (lon >= LONc) lon -= LONc;
  return (pl * LATc + lat) * LONc + lon;
}

// Load a 16x32 f16 fragment (A-style K-packed layout) from a row-major
// [rows][ld] matrix at tile origin p0. Works for global and LDS pointers.
// lane 0-15  : row = lane,    halves 0..7 = K0..7,  8..15 = K16..23
// lane 16-31 : row = lane-16, halves 0..7 = K8..15, 8..15 = K24..31
__device__ __forceinline__ v16h ld_frag(const _Float16* p0, int ld) {
  int lane = threadIdx.x & 31;
  int r    = lane & 15;
  int koff = (lane >> 4) << 3;            // 0 or 8
  const _Float16* p = p0 + (size_t)r * ld + koff;
  v8h lo = *(const v8h*)(p);
  v8h hi = *(const v8h*)(p + 16);
  v16h f;
#pragma unroll
  for (int i = 0; i < 8; ++i) { f[i] = lo[i]; f[i + 8] = hi[i]; }
  return f;
}

// ---------------- weight convert + transpose: w[K][N] f32 -> wT[N][K] f16 ----
__global__ void convert_transpose(const float* __restrict__ w,
                                  _Float16* __restrict__ wT, int K, int Nc) {
  int i = blockIdx.x * 256 + threadIdx.x;
  if (i < K * Nc) {
    int k = i / Nc, n = i - k * Nc;
    wT[(size_t)n * K + k] = (_Float16)w[i];
  }
}

// ---------------- LayerNorm (+ optional roll/window-partition gather) --------
// one wave per token, 6 channels per lane; writes f16 activations
template <bool PARTITION>
__launch_bounds__(256)
__global__ void ln_kernel(const float* __restrict__ x, const float* __restrict__ g,
                          const float* __restrict__ b, _Float16* __restrict__ out) {
  int t    = blockIdx.x * 8 + (threadIdx.x >> 5);
  int lane = threadIdx.x & 31;
  size_t src = PARTITION ? (size_t)win_to_orig(t) : (size_t)t;
  const float* xp = x + src * DIM;
  float vals[6]; float s = 0.f, s2 = 0.f;
#pragma unroll
  for (int j = 0; j < 6; ++j) {
    float v = xp[lane + 32 * j]; vals[j] = v; s += v; s2 += v * v;
  }
#pragma unroll
  for (int off = 16; off >= 1; off >>= 1) {
    s  += __shfl_xor(s,  off, 32);
    s2 += __shfl_xor(s2, off, 32);
  }
  float mean = s * (1.f / 192.f);
  float var  = s2 * (1.f / 192.f) - mean * mean;
  float rstd = rsqrtf(var + 1e-5f);
#pragma unroll
  for (int j = 0; j < 6; ++j) {
    int c = lane + 32 * j;
    out[(size_t)t * DIM + c] = (_Float16)((vals[j] - mean) * rstd * g[c] + b[c]);
  }
}

// ---------------- WMMA GEMM, async-LDS double-buffered, fused epilogues -----
// C[M x NN] = A[M x KK] * Bt[NN x KK]^T, M = 138240 (grid.x = 540 tiles of 256),
// block tile 256x64, 8 waves x (32x64), BK = 32, 2-deep async pipeline.
// KK/NN compile-time -> fully unrolled pipeline, immediate addressing.
// EPI 0: +bias -> f16 (QKV)
// EPI 1: +bias, window-reverse scatter, + shortcut x -> fp32 x1 (proj)
// EPI 2: +bias, exact GELU -> f16 (fc1)
// EPI 3: +bias, + x1 residual -> fp32 d_out (fc2)
template <int EPI, int KK, int NN>
__launch_bounds__(256)
__global__ void gemm_kernel(const _Float16* __restrict__ A,
                            const _Float16* __restrict__ Bt,
                            const float* __restrict__ bias,
                            void* __restrict__ outp,
                            const float* __restrict__ extra) {
  __shared__ __align__(16) _Float16 As[2][256 * 32];
  __shared__ __align__(16) _Float16 Bs[2][64 * 32];
  constexpr int NK = KK >> 5;
  const int tid  = threadIdx.x;
  const int lane = tid & 31;
  const int wave = tid >> 5;                    // 0..7 -> 32-row M strip
  const size_t rowBase = (size_t)blockIdx.x * 256;
  const int    colBase = blockIdx.y * 64;

  // per-thread async staging: thread tid owns A row `tid` (4 x b128) and one
  // B chunk (row tid>>2, 16B chunk tid&3) -> 5 async ops per tile.
  const _Float16* gA = A + (rowBase + tid) * KK;
  const _Float16* gB = Bt + (size_t)(colBase + (tid >> 2)) * KK + (tid & 3) * 8;

  auto stage = [&](int kb, int buf) {
    const _Float16* ga = gA + kb * 32;
    unsigned la = (unsigned)(uintptr_t)(&As[buf][tid * 32]);
#pragma unroll
    for (int u = 0; u < 4; ++u) async_ld128(la + u * 16, ga + u * 8);
    unsigned lb = (unsigned)(uintptr_t)(&Bs[buf][(tid >> 2) * 32 + (tid & 3) * 8]);
    async_ld128(lb, gB + kb * 32);
  };

  v8f acc[2][4] = {};
  stage(0, 0);
#pragma unroll
  for (int kb = 0; kb < NK; ++kb) {
    const int buf = kb & 1;
    if (kb + 1 < NK) {
      stage(kb + 1, buf ^ 1);
      wait_asynccnt<5>();       // first 5 (current tile) complete, in order
    } else {
      wait_asynccnt<0>();
    }
    __syncthreads();            // all waves' async LDS writes visible

    v16h a[2], b[4];
#pragma unroll
    for (int t = 0; t < 2; ++t)
      a[t] = ld_frag(&As[buf][(wave * 32 + t * 16) * 32], 32);
#pragma unroll
    for (int j = 0; j < 4; ++j)
      b[j] = ld_frag(&Bs[buf][(j * 16) * 32], 32);
#pragma unroll
    for (int t = 0; t < 2; ++t)
#pragma unroll
      for (int j = 0; j < 4; ++j)
        acc[t][j] = WMMA_F32_F16(a[t], b[j], acc[t][j]);

    __syncthreads();            // done reading buf before it is re-staged
  }

#pragma unroll
  for (int t = 0; t < 2; ++t)
#pragma unroll
    for (int j = 0; j < 4; ++j) {
      v8f a = acc[t][j];
#pragma unroll
      for (int e = 0; e < 8; ++e) {
        int m = e + ((lane >> 4) << 3);
        size_t row = rowBase + wave * 32 + t * 16 + m;
        int    col = colBase + j * 16 + (lane & 15);
        float v = a[e] + bias[col];
        if constexpr (EPI == 0) {
          ((_Float16*)outp)[row * (size_t)NN + col] = (_Float16)v;
        } else if constexpr (EPI == 1) {
          size_t orig = (size_t)win_to_orig((int)row);
          ((float*)outp)[orig * DIM + col] = extra[orig * DIM + col] + v;
        } else if constexpr (EPI == 2) {
          float gl = 0.5f * v * (1.0f + erff(v * 0.7071067811865476f));
          ((_Float16*)outp)[row * (size_t)NN + col] = (_Float16)gl;
        } else {
          ((float*)outp)[row * DIM + col] = v + extra[row * DIM + col];
        }
      }
    }
}

// ---------------- earth-specific window attention ----------------
// one block per (window, head); 9 waves, one 16-row query tile per wave.
// LDS: S (144x144 f32) | P (144x160 f16, zero-padded K) | VT (32x160 f16)
constexpr int SMEM_ATTN = Nt * Nt * 4 + Nt * 160 * 2 + HD * 160 * 2; // 139264 B

__launch_bounds__(288)
__global__ void attn_kernel(const _Float16* __restrict__ qkv,
                            const float* __restrict__ earth_bias,
                            _Float16* __restrict__ attn_out) {
  const int widx = blockIdx.x;          // 0..959
  const int h    = blockIdx.y;          // 0..5
  const int w    = widx % NWc;          // NW index (npl, nlat)
  const int npl  = w >> 4, nlat = w & 15;

  extern __shared__ char smem[];
  float*    S  = (float*)smem;                           // 144*144
  _Float16* P  = (_Float16*)(smem + Nt * Nt * 4);        // 144*160
  _Float16* VT = P + Nt * 160;                           // 32*160

  const int tid  = threadIdx.x;
  const int wave = tid >> 5;            // 0..8
  const int lane = tid & 31;

  const _Float16* qbase = qkv + (size_t)widx * Nt * (3 * DIM) + h * HD;
  const _Float16* kbase = qbase + DIM;
  const _Float16* vbase = qbase + 2 * DIM;

  // stage V^T (32 x 160, zero-padded keys) and zero P's pad columns
  for (int idx = tid; idx < HD * 160; idx += 288) {
    int d = idx / 160, n = idx - d * 160;
    VT[idx] = (n < Nt) ? vbase[(size_t)n * (3 * DIM) + d] : (_Float16)0.f;
  }
  for (int idx = tid; idx < Nt * 16; idx += 288) {
    int r = idx >> 4, c = Nt + (idx & 15);
    P[r * 160 + c] = (_Float16)0.f;
  }

  // ---- S = scale * q k^T + earth_bias + mask ----
  v16h aq = ld_frag(qbase + (size_t)(wave * 16) * (3 * DIM), 3 * DIM);
  for (int j = 0; j < 9; ++j) {
    v16h bk = ld_frag(kbase + (size_t)(j * 16) * (3 * DIM), 3 * DIM);
    v8f acc = {};
    acc = WMMA_F32_F16(aq, bk, acc);
#pragma unroll
    for (int e = 0; e < 8; ++e) {
      int m  = e + ((lane >> 4) << 3);
      int qi = wave * 16 + m;                    // query token in window
      int ki = j * 16 + (lane & 15);             // key token in window
      int wp1 = qi / 72, r1 = qi - wp1 * 72, wl1 = r1 / 12, ww1 = r1 - wl1 * 12;
      int wp2 = ki / 72, r2 = ki - wp2 * 72, wl2 = r2 / 12, ww2 = r2 - wl2 * 12;
      int bidx = 828 * (wp1 + 2 * wp2) + 23 * (wl1 + 6 * wl2) + (ww1 - ww2 + 11);
      float bias = earth_bias[((size_t)bidx * NWc + w) * HEADS + h];
      int plq = npl * 2 + wp1, latq = nlat * 6 + wl1;
      int plk = npl * 2 + wp2, latk = nlat * 6 + wl2;
      int rq = (plq < 6 ? 0 : (plq < 7 ? 1 : 2)) * 3 + (latq < 90 ? 0 : (latq < 93 ? 1 : 2));
      int rk = (plk < 6 ? 0 : (plk < 7 ? 1 : 2)) * 3 + (latk < 90 ? 0 : (latk < 93 ? 1 : 2));
      float val = acc[e] * SCALE + bias + ((rq != rk) ? -100.f : 0.f);
      S[qi * Nt + ki] = val;
    }
  }
  __syncthreads();

  // ---- softmax rows -> P (f16) ----
  for (int r = wave; r < Nt; r += 9) {
    float mx = -1e30f;
    for (int c = lane; c < Nt; c += 32) mx = fmaxf(mx, S[r * Nt + c]);
#pragma unroll
    for (int off = 16; off >= 1; off >>= 1) mx = fmaxf(mx, __shfl_xor(mx, off, 32));
    float sum = 0.f;
    for (int c = lane; c < Nt; c += 32) {
      float ev = __expf(S[r * Nt + c] - mx);
      S[r * Nt + c] = ev;
      sum += ev;
    }
#pragma unroll
    for (int off = 16; off >= 1; off >>= 1) sum += __shfl_xor(sum, off, 32);
    float inv = 1.f / sum;
    for (int c = lane; c < Nt; c += 32)
      P[r * 160 + c] = (_Float16)(S[r * Nt + c] * inv);
  }
  __syncthreads();

  // ---- O = P @ V^T  (K padded to 160 = 5 WMMA steps) ----
  v8f o0 = {}, o1 = {};
#pragma unroll
  for (int kk = 0; kk < 5; ++kk) {
    v16h pa = ld_frag(P + (size_t)(wave * 16) * 160 + kk * 32, 160);
    v16h b0 = ld_frag(VT + 0 * 160 + kk * 32, 160);
    v16h b1 = ld_frag(VT + (size_t)16 * 160 + kk * 32, 160);
    o0 = WMMA_F32_F16(pa, b0, o0);
    o1 = WMMA_F32_F16(pa, b1, o1);
  }
#pragma unroll
  for (int e = 0; e < 8; ++e) {
    int m = e + ((lane >> 4) << 3);
    size_t tok = (size_t)widx * Nt + wave * 16 + m;
    attn_out[tok * DIM + h * HD + (lane & 15)]      = (_Float16)o0[e];
    attn_out[tok * DIM + h * HD + 16 + (lane & 15)] = (_Float16)o1[e];
  }
}

// ---------------- host launcher ----------------
extern "C" void kernel_launch(void* const* d_in, const int* in_sizes, int n_in,
                              void* d_out, int out_size, void* d_ws, size_t ws_size,
                              hipStream_t stream) {
  (void)in_sizes; (void)n_in; (void)out_size; (void)ws_size;
  const float* x      = (const float*)d_in[0];
  const float* n1g    = (const float*)d_in[1];
  const float* n1b    = (const float*)d_in[2];
  const float* qkv_w  = (const float*)d_in[3];
  const float* qkv_b  = (const float*)d_in[4];
  const float* ebias  = (const float*)d_in[5];
  const float* proj_w = (const float*)d_in[6];
  const float* proj_b = (const float*)d_in[7];
  const float* n2g    = (const float*)d_in[8];
  const float* n2b    = (const float*)d_in[9];
  const float* fc1_w  = (const float*)d_in[10];
  const float* fc1_b  = (const float*)d_in[11];
  const float* fc2_w  = (const float*)d_in[12];
  const float* fc2_b  = (const float*)d_in[13];

  char* ws = (char*)d_ws;
  _Float16* xw    = (_Float16*)(ws);             // xw (f16); later reused as attn_out
  _Float16* qkvh  = (_Float16*)(ws + OFF_QKV);   // qkv (f16); later reused as h2
  float*    x1    = (float*)   (ws + OFF_X1);    // attn residual (fp32)
  _Float16* act   = (_Float16*)(ws + OFF_ACT);   // mlp hidden (f16)
  _Float16* wqkvT = (_Float16*)(ws + OFF_W);
  _Float16* wprjT = wqkvT + (size_t)3 * DIM * DIM;
  _Float16* wfc1T = wprjT + (size_t)DIM * DIM;
  _Float16* wfc2T = wfc1T + (size_t)4 * DIM * DIM;

  // weights -> f16 [N][K]
  convert_transpose<<<(DIM * 3 * DIM + 255) / 256, 256, 0, stream>>>(qkv_w, wqkvT, DIM, 3 * DIM);
  convert_transpose<<<(DIM * DIM + 255) / 256,     256, 0, stream>>>(proj_w, wprjT, DIM, DIM);
  convert_transpose<<<(DIM * 4 * DIM + 255) / 256, 256, 0, stream>>>(fc1_w, wfc1T, DIM, 4 * DIM);
  convert_transpose<<<(4 * DIM * DIM + 255) / 256, 256, 0, stream>>>(fc2_w, wfc2T, 4 * DIM, DIM);

  // LN1 + roll + window partition -> xw (f16, window token order)
  ln_kernel<true><<<LTOK / 8, 256, 0, stream>>>(x, n1g, n1b, xw);

  // QKV: [138240 x 192] @ [192 x 576] -> qkv f16
  gemm_kernel<0, 192, 576><<<dim3(540, 9), 256, 0, stream>>>(xw, wqkvT, qkv_b, qkvh, nullptr);

  // windowed attention -> attn_out f16 (reuses xw buffer)
  attn_kernel<<<dim3(NWIN, HEADS), 288, SMEM_ATTN, stream>>>(qkvh, ebias, xw);

  // proj + window reverse + roll back + shortcut -> x1 fp32
  gemm_kernel<1, 192, 192><<<dim3(540, 3), 256, 0, stream>>>(xw, wprjT, proj_b, x1, x);

  // LN2 -> h2 f16 (reuses qkv buffer)
  ln_kernel<false><<<LTOK / 8, 256, 0, stream>>>(x1, n2g, n2b, qkvh);

  // fc1 + exact GELU -> act f16
  gemm_kernel<2, 192, 768><<<dim3(540, 12), 256, 0, stream>>>(qkvh, wfc1T, fc1_b, act, nullptr);

  // fc2 + residual -> d_out fp32
  gemm_kernel<3, 768, 192><<<dim3(540, 3), 256, 0, stream>>>(act, wfc2T, fc2_b, d_out, x1);
}
